// Attention_23330262352360
// MI455X (gfx1250) — matrix operand order repaired and checked by reading, not compiled, not run on hardware
//
#include <hip/hip_runtime.h>

#define HEADS   8
#define DK      64
#define DIMIN   768
#define NSEQ    2048
#define BATCH   2
#define FREL    192
#define NPOS    4095      // 2*NSEQ-1
#define HD      512       // HEADS*DK
#define ATT_WAVES 8

typedef __attribute__((ext_vector_type(16))) _Float16 v16h;
typedef __attribute__((ext_vector_type(8)))  _Float16 v8h;
typedef __attribute__((ext_vector_type(8)))  float    v8f;

union Frag { v16h v; v8h h[2]; };

__device__ __forceinline__ v8f wmma_f16(const Frag& a, const Frag& b, v8f c) {
  return __builtin_amdgcn_wmma_f32_16x16x32_f16(false, a.v, false, b.v, (short)0, c, false, false);
}
__device__ __forceinline__ v8h ldg8(const _Float16* p) { return *(const v8h*)p; }

// ---------------------------------------------------------------------------
// Positional embedding: pos[4095, 192] (f16). One thread per position row.
// ---------------------------------------------------------------------------
__global__ void embed_kernel(_Float16* __restrict__ pos) {
  int p = blockIdx.x * blockDim.x + threadIdx.x;
  if (p >= NPOS) return;
  float dist = (float)(p - (NSEQ - 1));
  float ad   = fabsf(dist);
  float sgn  = dist > 0.f ? 1.f : (dist < 0.f ? -1.f : 0.f);

  float vals[96];
  // exponential basis: half_life = 2^linspace(3, log2(2048)=11, 32)
  #pragma unroll 4
  for (int c = 0; c < 32; ++c) {
    float hlife = exp2f(3.f + 8.f * (float)c / 31.f);
    vals[c] = exp2f(-ad / hlife);
  }
  // central mask: widths = 2^(c+1) - 1
  #pragma unroll 4
  for (int c = 0; c < 32; ++c) {
    float width = exp2f((float)(c + 1)) - 1.f;
    vals[32 + c] = (width > ad) ? 1.f : 0.f;
  }
  // gamma pdf (double to tame lgamma cancellation), normalized by row max
  double probs[32]; double mx = 0.0;
  double adl = (double)ad;
  for (int c = 0; c < 32; ++c) {
    double mean = 64.0 * (double)(c + 1);      // linspace(64, 2048, 32)
    double t    = mean / 32.0;                 // stddev = 32
    double conc = t * t;
    double rate = mean / 1024.0;               // mean / stddev^2
    double lu   = (conc - 1.0) * log(adl) - rate * adl;   // xlogy; ad=0 -> -inf
    double ln_  = lgamma(conc) - conc * log(rate);
    double pr   = exp(lu - ln_) + 1e-8;
    probs[c] = pr; if (pr > mx) mx = pr;
  }
  for (int c = 0; c < 32; ++c) vals[64 + c] = (float)(probs[c] / mx);

  _Float16* row = pos + (size_t)p * FREL;
  for (int c = 0; c < 96; ++c) {
    row[c]      = (_Float16)vals[c];
    row[96 + c] = (_Float16)(sgn * vals[c]);
  }
}

// ---------------------------------------------------------------------------
// f32 -> f16 convert, optionally transposed (weights become [N,K] row-major).
// ---------------------------------------------------------------------------
__global__ void cvt_kernel(const float* __restrict__ src, _Float16* __restrict__ dst,
                           int rows, int cols, int transpose) {
  int i = blockIdx.x * blockDim.x + threadIdx.x;
  if (i >= rows * cols) return;
  int r = i / cols, c = i % cols;
  float v = src[i];
  if (transpose) dst[(size_t)c * rows + r] = (_Float16)v;
  else           dst[i] = (_Float16)v;
}

// ---------------------------------------------------------------------------
// Generic WMMA GEMM: C[M,N] = alpha * A[M,K] * Bt[N,K]^T + bias[N]
// mode 0: f16 out; mode 1: f32 out; mode 2: f16 out transposed per batch:
//         C[(row/tmb)*N*tmb + col*tmb + row%tmb]   (for V^T storage)
// One 16x16 C tile per wave.  K must be a multiple of 32, N of 16.
// ---------------------------------------------------------------------------
__global__ __launch_bounds__(256)
void gemm_wmma(const _Float16* __restrict__ A, const _Float16* __restrict__ Bt,
               void* __restrict__ C, const float* __restrict__ bias,
               int M, int N, int K, float alpha, int mode, int tmb) {
  const int lane = threadIdx.x & 31;
  const int sub  = lane >> 4;          // lane half: 0 or 1
  const int l16  = lane & 15;
  const int wid  = blockIdx.x * (blockDim.x >> 5) + (threadIdx.x >> 5);
  const int tilesN = N >> 4;
  const int tilesM = (M + 15) >> 4;
  if (wid >= tilesM * tilesN) return;        // wave-uniform exit (EXEC stays all-1)
  const int m0 = (wid / tilesN) << 4;
  const int n0 = (wid % tilesN) << 4;

  int ar = m0 + l16; if (ar > M - 1) ar = M - 1;        // clamp (dup row, unused)
  const _Float16* ap = A  + (size_t)ar * K;
  const _Float16* bp = Bt + (size_t)(n0 + l16) * K;

  v8f acc = {};
  for (int k0 = 0; k0 < K; k0 += 32) {
    Frag a, b;
    a.h[0] = ldg8(ap + k0 + sub * 16);       // A: contiguous K = 16*sub .. +15
    a.h[1] = ldg8(ap + k0 + sub * 16 + 8);
    b.h[0] = ldg8(bp + k0 + sub * 16);       // B: contiguous K = 16*sub .. +15
    b.h[1] = ldg8(bp + k0 + sub * 16 + 8);
    acc = wmma_f16(a, b, acc);
  }
  #pragma unroll
  for (int r = 0; r < 8; ++r) {
    int row = m0 + r + 8 * sub;
    if (row >= M) continue;
    int col = n0 + l16;
    float v = acc[r] * alpha + (bias ? bias[col] : 0.f);
    if (mode == 0)      ((_Float16*)C)[(size_t)row * N + col] = (_Float16)v;
    else if (mode == 1) ((float*)C)[(size_t)row * N + col] = v;
    else {
      int bb = row / tmb, mm = row - bb * tmb;
      ((_Float16*)C)[((size_t)bb * N + col) * (size_t)tmb + mm] = (_Float16)v;
    }
  }
}

// ---------------------------------------------------------------------------
// Fused relative-position flash attention.
//   qc,qr,kk : [B*N, 512] f16   (q already scaled + biased)
//   vT       : [B, 512, N] f16  (transposed V)
//   relk     : [4095, 512] f16
//   outp     : [B*N, 512] f16
// 8 waves/block, one 16-row query tile per wave, 32 keys per step.
// ---------------------------------------------------------------------------
__global__ __launch_bounds__(256)
void attn_kernel(const _Float16* __restrict__ qc, const _Float16* __restrict__ qr,
                 const _Float16* __restrict__ kk, const _Float16* __restrict__ vT,
                 const _Float16* __restrict__ relk, _Float16* __restrict__ outp) {
  __shared__ alignas(16) _Float16 lds_p[ATT_WAVES][16 * 40];  // padded P tiles

  const int lane = threadIdx.x & 31;
  const int sub  = lane >> 4;
  const int l16  = lane & 15;
  const int w    = threadIdx.x >> 5;
  const int blocksPerBH = NSEQ / (16 * ATT_WAVES);            // 16
  const int bh = blockIdx.x / blocksPerBH;
  const int b  = bh >> 3, h = bh & 7;
  const int i0 = (blockIdx.x % blocksPerBH) * (16 * ATT_WAVES) + w * 16;

  // Persistent A fragments for q_c and q_r (K = 64 -> two 16x32 frags each)
  const _Float16* qcp = qc + (size_t)(b * NSEQ + i0 + l16) * HD + h * DK;
  const _Float16* qrp = qr + (size_t)(b * NSEQ + i0 + l16) * HD + h * DK;
  Frag aqc0, aqc1, aqr0, aqr1;
  aqc0.h[0] = ldg8(qcp + sub * 16);      aqc0.h[1] = ldg8(qcp + sub * 16 + 8);
  aqc1.h[0] = ldg8(qcp + 32 + sub * 16); aqc1.h[1] = ldg8(qcp + 32 + sub * 16 + 8);
  aqr0.h[0] = ldg8(qrp + sub * 16);      aqr0.h[1] = ldg8(qrp + sub * 16 + 8);
  aqr1.h[0] = ldg8(qrp + 32 + sub * 16); aqr1.h[1] = ldg8(qrp + 32 + sub * 16 + 8);

  v8f o0 = {}, o1 = {}, o2 = {}, o3 = {};
  float mrow[8], lrow[8];
  #pragma unroll
  for (int r = 0; r < 8; ++r) { mrow[r] = -1e30f; lrow[r] = 0.f; }

  const _Float16* kbase = kk   + (size_t)(b * NSEQ) * HD + h * DK;
  const _Float16* vbase = vT   + (size_t)(b * HD + h * DK) * NSEQ;
  const _Float16* rbase = relk + h * DK;
  _Float16* pl = lds_p[w];

  #pragma unroll 1
  for (int j0 = 0; j0 < NSEQ; j0 += 32) {
    // ---- warm WGP$/L2 one step ahead (global_prefetch_b8; no counter cost) ----
    if (j0 + 32 < NSEQ) {
      __builtin_prefetch(kbase + (size_t)(j0 + 32 + l16) * HD, 0, 3);        // next K rows
      __builtin_prefetch(kbase + (size_t)(j0 + 48 + l16) * HD, 0, 3);
      __builtin_prefetch(vbase + (size_t)l16 * NSEQ + j0 + 32, 0, 3);        // next V rows
      __builtin_prefetch(vbase + (size_t)(32 + l16) * NSEQ + j0 + 32, 0, 3);
      int dn = (NSEQ - 1) + (j0 + 32) - i0 - 15 + l16;                       // next rel-k rows
      dn = dn < 0 ? 0 : (dn > NPOS - 1 ? NPOS - 1 : dn);
      __builtin_prefetch(rbase + (size_t)dn * HD, 0, 3);
    }

    v8f s[2];
    #pragma unroll
    for (int jt = 0; jt < 2; ++jt) {
      const int jb = j0 + jt * 16;
      // ---- content logits: q_c @ k^T ----
      const _Float16* kp = kbase + (size_t)(jb + l16) * HD;
      Frag bk;
      v8f sc = {};
      bk.h[0] = ldg8(kp + sub * 16);      bk.h[1] = ldg8(kp + sub * 16 + 8);
      sc = wmma_f16(aqc0, bk, sc);
      bk.h[0] = ldg8(kp + 32 + sub * 16); bk.h[1] = ldg8(kp + 32 + sub * 16 + 8);
      sc = wmma_f16(aqc1, bk, sc);

      // ---- rel logits: T[16,32] = q_r @ relk[base-15 .. base+16]^T ----
      const int base = (NSEQ - 1) + jb - i0;
      int d0 = base - 15 + l16; d0 = d0 < 0 ? 0 : (d0 > NPOS - 1 ? NPOS - 1 : d0);
      int d1 = base + 1  + l16; d1 = d1 < 0 ? 0 : (d1 > NPOS - 1 ? NPOS - 1 : d1);
      const _Float16* r0p = rbase + (size_t)d0 * HD;
      const _Float16* r1p = rbase + (size_t)d1 * HD;
      Frag br;
      v8f t0 = {}, t1 = {};
      br.h[0] = ldg8(r0p + sub * 16);      br.h[1] = ldg8(r0p + sub * 16 + 8);
      t0 = wmma_f16(aqr0, br, t0);
      br.h[0] = ldg8(r0p + 32 + sub * 16); br.h[1] = ldg8(r0p + 32 + sub * 16 + 8);
      t0 = wmma_f16(aqr1, br, t0);
      br.h[0] = ldg8(r1p + sub * 16);      br.h[1] = ldg8(r1p + sub * 16 + 8);
      t1 = wmma_f16(aqr0, br, t1);
      br.h[0] = ldg8(r1p + 32 + sub * 16); br.h[1] = ldg8(r1p + 32 + sub * 16 + 8);
      t1 = wmma_f16(aqr1, br, t1);

      // ---- fused relative_shift: s(row,col) += T[row, col - row + 15] ----
      const int hb = sub * 16;
      #pragma unroll
      for (int r = 0; r < 8; ++r) {
        int row = r + 8 * sub;
        int c2  = l16 - row + 15;            // in [0,30]
        int src = hb + (c2 & 15);            // same half -> same row set
        float va = __shfl(t0[r], src, 32);
        float vb = __shfl(t1[r], src, 32);
        sc[r] += (c2 < 16) ? va : vb;
      }
      s[jt] = sc;
    }

    // ---- online softmax over this 32-key step ----
    float al[8];
    #pragma unroll
    for (int r = 0; r < 8; ++r) {
      float mx = fmaxf(s[0][r], s[1][r]);
      mx = fmaxf(mx, __shfl_xor(mx, 1, 16));
      mx = fmaxf(mx, __shfl_xor(mx, 2, 16));
      mx = fmaxf(mx, __shfl_xor(mx, 4, 16));
      mx = fmaxf(mx, __shfl_xor(mx, 8, 16));
      float mn = fmaxf(mrow[r], mx);
      float a_ = __expf(mrow[r] - mn);
      float p0 = __expf(s[0][r] - mn);
      float p1 = __expf(s[1][r] - mn);
      float sm = p0 + p1;
      sm += __shfl_xor(sm, 1, 16);
      sm += __shfl_xor(sm, 2, 16);
      sm += __shfl_xor(sm, 4, 16);
      sm += __shfl_xor(sm, 8, 16);
      lrow[r] = lrow[r] * a_ + sm;
      mrow[r] = mn;
      al[r] = a_;
      s[0][r] = p0; s[1][r] = p1;
    }
    #pragma unroll
    for (int r = 0; r < 8; ++r) {
      o0[r] *= al[r]; o1[r] *= al[r]; o2[r] *= al[r]; o3[r] *= al[r];
    }

    // ---- P (C-layout) -> LDS -> A-layout fragment ----
    #pragma unroll
    for (int r = 0; r < 8; ++r) {
      int row = r + 8 * sub;
      pl[row * 40 + l16]      = (_Float16)s[0][r];
      pl[row * 40 + 16 + l16] = (_Float16)s[1][r];
    }
    Frag pf;   // same-wave LDS ops are in-order; compiler inserts dscnt waits
    pf.h[0] = *(const v8h*)(pl + l16 * 40 + sub * 16);
    pf.h[1] = *(const v8h*)(pl + l16 * 40 + sub * 16 + 8);

    // ---- O += P @ V  (4 dv tiles of 16) ----
    Frag bv;
    const _Float16* vp = vbase + (size_t)l16 * NSEQ + j0;
    bv.h[0] = ldg8(vp + sub * 16);               bv.h[1] = ldg8(vp + sub * 16 + 8);
    o0 = wmma_f16(pf, bv, o0);
    bv.h[0] = ldg8(vp + 16 * NSEQ + sub * 16);   bv.h[1] = ldg8(vp + 16 * NSEQ + sub * 16 + 8);
    o1 = wmma_f16(pf, bv, o1);
    bv.h[0] = ldg8(vp + 32 * NSEQ + sub * 16);   bv.h[1] = ldg8(vp + 32 * NSEQ + sub * 16 + 8);
    o2 = wmma_f16(pf, bv, o2);
    bv.h[0] = ldg8(vp + 48 * NSEQ + sub * 16);   bv.h[1] = ldg8(vp + 48 * NSEQ + sub * 16 + 8);
    o3 = wmma_f16(pf, bv, o3);
  }

  // ---- epilogue: normalize and store f16 [b*n, h*64+d] ----
  #pragma unroll
  for (int r = 0; r < 8; ++r) {
    int row = r + 8 * sub;
    float inv = 1.f / lrow[r];
    _Float16* op = outp + (size_t)(b * NSEQ + i0 + row) * HD + h * DK + l16;
    op[0]  = (_Float16)(o0[r] * inv);
    op[16] = (_Float16)(o1[r] * inv);
    op[32] = (_Float16)(o2[r] * inv);
    op[48] = (_Float16)(o3[r] * inv);
  }
}

// ---------------------------------------------------------------------------
extern "C" void kernel_launch(void* const* d_in, const int* in_sizes, int n_in,
                              void* d_out, int out_size, void* d_ws, size_t ws_size,
                              hipStream_t stream) {
  (void)in_sizes; (void)n_in; (void)out_size; (void)ws_size;
  const float* x     = (const float*)d_in[0];   // [2,2048,768]
  const float* Wq    = (const float*)d_in[1];   // [768,512]
  const float* Wk    = (const float*)d_in[2];
  const float* Wv    = (const float*)d_in[3];
  const float* Wrel  = (const float*)d_in[4];   // [192,512]
  const float* Wout  = (const float*)d_in[5];   // [512,768]
  const float* b_out = (const float*)d_in[6];   // [768]
  const float* bc    = (const float*)d_in[7];   // [1,8,1,64] -> flat [512]
  const float* bp    = (const float*)d_in[8];   // [1,8,1,64] -> flat [512]
  float* out = (float*)d_out;

  char* ws = (char*)d_ws;
  size_t off = 0;
  auto carve = [&](size_t halves) -> _Float16* {
    off = (off + 255) & ~(size_t)255;
    _Float16* p = (_Float16*)(ws + off);
    off += halves * sizeof(_Float16);
    return p;
  };
  const int M = BATCH * NSEQ;                       // 4096
  _Float16* pos   = carve((size_t)NPOS * FREL);
  _Float16* xh    = carve((size_t)M * DIMIN);
  _Float16* wqt   = carve((size_t)HD * DIMIN);
  _Float16* wkt   = carve((size_t)HD * DIMIN);
  _Float16* wvt   = carve((size_t)HD * DIMIN);
  _Float16* wrelt = carve((size_t)HD * FREL);
  _Float16* woutt = carve((size_t)DIMIN * HD);
  _Float16* qcb   = carve((size_t)M * HD);
  _Float16* qrb   = carve((size_t)M * HD);
  _Float16* kb    = carve((size_t)M * HD);
  _Float16* vTb   = carve((size_t)BATCH * HD * NSEQ);
  _Float16* relkb = carve((size_t)NPOS * HD);
  _Float16* aob   = carve((size_t)M * HD);

  auto cvt = [&](const float* s, _Float16* d, int r, int c, int t) {
    int n = r * c;
    cvt_kernel<<<(n + 255) / 256, 256, 0, stream>>>(s, d, r, c, t);
  };
  auto gemm = [&](const _Float16* A, const _Float16* Bt, void* C, const float* bias,
                  int Mm, int Nn, int Kk, float alpha, int mode, int tmb) {
    int tiles = ((Mm + 15) / 16) * (Nn / 16);
    gemm_wmma<<<(tiles + 7) / 8, 256, 0, stream>>>(A, Bt, C, bias, Mm, Nn, Kk, alpha, mode, tmb);
  };

  // 1) positional embeddings
  embed_kernel<<<(NPOS + 63) / 64, 64, 0, stream>>>(pos);
  // 2) precision conversion (+ weight transpose to [N,K])
  cvt(x,    xh,    M,      DIMIN, 0);
  cvt(Wq,   wqt,   DIMIN,  HD,    1);
  cvt(Wk,   wkt,   DIMIN,  HD,    1);
  cvt(Wv,   wvt,   DIMIN,  HD,    1);
  cvt(Wrel, wrelt, FREL,   HD,    1);
  cvt(Wout, woutt, HD,     DIMIN, 1);
  // 3) projections (scale + biases folded into q variants)
  const float scale = 0.125f;                       // dk^-0.5
  gemm(xh,  wqt,   qcb,   bc,    M,    HD, DIMIN, scale, 0, 0);   // q_c
  gemm(xh,  wqt,   qrb,   bp,    M,    HD, DIMIN, scale, 0, 0);   // q_r
  gemm(xh,  wkt,   kb,    nullptr, M,  HD, DIMIN, 1.f,   0, 0);   // k
  gemm(xh,  wvt,   vTb,   nullptr, M,  HD, DIMIN, 1.f,   2, NSEQ);// v (transposed)
  gemm(pos, wrelt, relkb, nullptr, NPOS, HD, FREL, 1.f,  0, 0);   // rel_k
  // 4) fused relative-position flash attention
  attn_kernel<<<BATCH * HEADS * (NSEQ / (16 * ATT_WAVES)), 32 * ATT_WAVES, 0, stream>>>(
      qcb, qrb, kb, vTb, relkb, aob);
  // 5) output projection (f32 out + bias)
  gemm(aob, woutt, out, b_out, M, DIMIN, HD, 1.f, 1, 0);
}
